// LowRankSelfAttention_33749853012055
// MI455X (gfx1250) — compile-verified, matrix operand order checked by reading, not run on hardware
//
#include <hip/hip_runtime.h>
#include <hip/hip_bf16.h>

// ---------------- problem constants ----------------
#define S_LEN 4096
#define HID   2048
#define HEADS 16
#define DH    128
#define RANK  1024

typedef __bf16 bf16_t;
typedef __attribute__((ext_vector_type(16))) __bf16 v16bf;
typedef __attribute__((ext_vector_type(8)))  __bf16 v8bf;
typedef __attribute__((ext_vector_type(8)))  float  v8f;
typedef __attribute__((ext_vector_type(4)))  int    v4i;

// ---------------- CDNA5 async global->LDS path (guarded) ----------------
#if defined(__gfx1250__) && __has_builtin(__builtin_amdgcn_global_load_async_to_lds_b128)
#define HAVE_ASYNC_LDS 1
#else
#define HAVE_ASYNC_LDS 0
#endif

__device__ __forceinline__ void async_copy16(const void* gsrc, void* ldst) {
#if HAVE_ASYNC_LDS
    // builtin signature: (v4i addrspace(1)*, v4i addrspace(3)*, imm offset, imm cpol)
    __builtin_amdgcn_global_load_async_to_lds_b128(
        (__attribute__((address_space(1))) v4i*)gsrc,
        (__attribute__((address_space(3))) v4i*)ldst,
        /*offset=*/0, /*cpol=*/0);
#else
    *(uint4*)ldst = *(const uint4*)gsrc;
#endif
}

template <int N>
__device__ __forceinline__ void wait_async() {
#if HAVE_ASYNC_LDS
#if __has_builtin(__builtin_amdgcn_s_wait_asynccnt)
    __builtin_amdgcn_s_wait_asynccnt(N);
#else
    if (N == 0) asm volatile("s_wait_asynccnt 0x0" ::: "memory");
    else        asm volatile("s_wait_asynccnt 0x8" ::: "memory");
#endif
#endif
}

__device__ __forceinline__ v8f vzero8() {
    v8f z = {0.f, 0.f, 0.f, 0.f, 0.f, 0.f, 0.f, 0.f};
    return z;
}

__device__ __forceinline__ v8f wmma_bf16f32(v16bf a, v16bf b, v8f c) {
    // D = A(16x32 bf16) * B(32x16 bf16) + C(16x16 f32)
    return __builtin_amdgcn_wmma_f32_16x16x32_bf16(
        /*neg_a=*/false, a, /*neg_b=*/false, b,
        /*c_mod=*/(short)0, c, /*reuse_a=*/false, /*reuse_b=*/false);
}

// Load one 16x32 bf16 fragment (A-style or B-style) from row-major storage:
// lane%16 selects the row (M for A, N for B); K contiguous along the row.
// Lane half h holds K chunks [h*8, h*8+8) and [16+h*8, 16+h*8+8).
__device__ __forceinline__ v16bf frag_ld(const bf16_t* base, int ld, int lane) {
    const int r  = lane & 15;
    const int hh = lane >> 4;
    const bf16_t* p = base + r * ld + hh * 8;
    v16bf f;
    ((v8bf*)&f)[0] = *(const v8bf*)(p);
    ((v8bf*)&f)[1] = *(const v8bf*)(p + 16);
    return f;
}

// ---------------- fp32 -> bf16 convert ----------------
__global__ __launch_bounds__(256)
void conv_f32_bf16(const float* __restrict__ src, bf16_t* __restrict__ dst, int n) {
    int i = (blockIdx.x * blockDim.x + threadIdx.x) * 4;
    if (i + 3 < n) {
        float4 v = *(const float4*)(src + i);
        dst[i + 0] = (bf16_t)v.x;
        dst[i + 1] = (bf16_t)v.y;
        dst[i + 2] = (bf16_t)v.z;
        dst[i + 3] = (bf16_t)v.w;
    } else {
        for (int t = i; t < n; ++t) dst[t] = (bf16_t)src[t];
    }
}

// ---------------- tiled bf16 WMMA GEMM (double-buffered async staging) ----
// C[M,N] = A[M,K] * B[N,K]^T  (both row-major, K contiguous)
// Block tile 128x128, 4 waves, each wave a 64x64 tile (4x4 v8f accs).
template <bool OUT_BF16>
__global__ __launch_bounds__(128)
void gemm_bf16_wmma(const bf16_t* __restrict__ A, const bf16_t* __restrict__ B,
                    void* __restrict__ C, const float* __restrict__ bias,
                    int M, int N, int K) {
    constexpr int LDT = 40; // 32 + 8 pad -> conflict-free 16B ds reads
    __shared__ bf16_t As[2][128 * LDT];
    __shared__ bf16_t Bs[2][128 * LDT];

    const int tid  = threadIdx.x;
    const int wave = tid >> 5;
    const int lane = tid & 31;
    const int wm = (wave >> 1) * 64;
    const int wn = (wave & 1) * 64;
    const int m0 = blockIdx.y * 128;
    const int n0 = blockIdx.x * 128;
    (void)M;

    v8f acc[4][4];
#pragma unroll
    for (int i = 0; i < 4; ++i)
#pragma unroll
        for (int j = 0; j < 4; ++j) acc[i][j] = vzero8();

    // per-thread: one 32-element row slice of A and of B -> 8 x 16B copies
    auto stage = [&](int buf, int k0) {
        const bf16_t* ga = A + (size_t)(m0 + tid) * K + k0;
        const bf16_t* gb = B + (size_t)(n0 + tid) * K + k0;
        bf16_t* la = &As[buf][tid * LDT];
        bf16_t* lb = &Bs[buf][tid * LDT];
#pragma unroll
        for (int t = 0; t < 4; ++t) async_copy16(ga + t * 8, la + t * 8);
#pragma unroll
        for (int t = 0; t < 4; ++t) async_copy16(gb + t * 8, lb + t * 8);
    };

    const int nk = K / 32;
    stage(0, 0);
    for (int ik = 0; ik < nk; ++ik) {
        const int buf = ik & 1;
        const bool has_next = (ik + 1) < nk;
        if (has_next) stage(buf ^ 1, (ik + 1) * 32);
        // current tile complete when only the next tile's 8 copies are pending
        if (has_next) wait_async<8>(); else wait_async<0>();
        __syncthreads();

        v16bf af[4], bfr[4];
#pragma unroll
        for (int i = 0; i < 4; ++i)
            af[i] = frag_ld(&As[buf][(wm + i * 16) * LDT], LDT, lane);
#pragma unroll
        for (int j = 0; j < 4; ++j)
            bfr[j] = frag_ld(&Bs[buf][(wn + j * 16) * LDT], LDT, lane);
#pragma unroll
        for (int i = 0; i < 4; ++i)
#pragma unroll
            for (int j = 0; j < 4; ++j)
                acc[i][j] = wmma_bf16f32(af[i], bfr[j], acc[i][j]);

        __syncthreads(); // all reads of buf done before it is restaged
    }

    // C/D layout: elem e of v8f at lane L -> row = e + 8*(L>>4), col = L&15
    const int cr = (lane >> 4) * 8;
    const int cc = lane & 15;
#pragma unroll
    for (int i = 0; i < 4; ++i)
#pragma unroll
        for (int j = 0; j < 4; ++j) {
            const int mb = m0 + wm + i * 16 + cr;
            const int nn = n0 + wn + j * 16 + cc;
#pragma unroll
            for (int e = 0; e < 8; ++e) {
                float v = acc[i][j][e];
                if constexpr (OUT_BF16) {
                    ((bf16_t*)C)[(size_t)(mb + e) * N + nn] = (bf16_t)v;
                } else {
                    ((float*)C)[(size_t)(mb + e) * N + nn] = v + bias[nn];
                }
            }
        }
}

// ---------------- flash attention (bf16 WMMA, online softmax) ----------------
// grid = (S/64 q-blocks, HEADS); block = 128 threads = 4 waves.
// Each wave owns 16 q-rows x full 128-d output (8 v8f accumulators).
// K tile and mask tile staged with async global->LDS; V transposed manually.
__global__ __launch_bounds__(128)
void flash_attn_wmma(const bf16_t* __restrict__ Q, const bf16_t* __restrict__ Kg,
                     const bf16_t* __restrict__ Vg, const float* __restrict__ mask,
                     bf16_t* __restrict__ O) {
    constexpr int LDK = DH + 8; // 136
    constexpr int LDV = 64 + 8; // 72
    constexpr int LDM = 68;     // mask row stride (fp32)
    __shared__ bf16_t Ks[64 * LDK];       // K block: [key][d]
    __shared__ bf16_t Vts[DH * LDV];      // V block transposed: [d][key]
    __shared__ bf16_t Ps[4 * 16 * LDV];   // per-wave P: [qrow][key]
    __shared__ float  Ms[64 * LDM];       // mask tile: [qrow][key] fp32

    const int head = blockIdx.y;
    const int qb   = blockIdx.x;
    const int tid  = threadIdx.x;
    const int wave = tid >> 5;
    const int lane = tid & 31;
    const int hh = lane >> 4;
    const int ln = lane & 15;
    const int q0 = qb * 64 + wave * 16;
    const size_t hoff = (size_t)head * DH;

    // Q fragments held in registers for the whole kernel (16 rows x 128 d)
    v16bf qf[4];
#pragma unroll
    for (int kc = 0; kc < 4; ++kc)
        qf[kc] = frag_ld(Q + (size_t)q0 * HID + hoff + kc * 32, HID, lane);

    v8f oacc[8];
#pragma unroll
    for (int nf = 0; nf < 8; ++nf) oacc[nf] = vzero8();
    float mrow[8], lrow[8];
#pragma unroll
    for (int j = 0; j < 8; ++j) { mrow[j] = -3.0e38f; lrow[j] = 0.f; }

    bf16_t* Pw = Ps + wave * 16 * LDV;
    const float kscale = 0.08838834764831845f;  // 1/sqrt(128)
    const float log2e  = 1.4426950408889634f;

    for (int kb = 0; kb < S_LEN / 64; ++kb) {
        const int k0 = kb * 64;
        __syncthreads(); // previous iteration readers done

        { // async stage K block (row-major, d contiguous): thread -> half row
            const int r = tid >> 1, c = (tid & 1) * 64;
            const bf16_t* src = Kg + (size_t)(k0 + r) * HID + hoff + c;
            bf16_t* dst = Ks + r * LDK + c;
#pragma unroll
            for (int t = 0; t < 8; ++t) async_copy16(src + t * 8, dst + t * 8);
        }
        { // async stage mask tile (64 x 64 fp32): thread -> half row
            const int r = tid >> 1, c = (tid & 1) * 32;
            const float* src = mask + (size_t)(qb * 64 + r) * S_LEN + k0 + c;
            float* dst = Ms + r * LDM + c;
#pragma unroll
            for (int t = 0; t < 8; ++t) async_copy16(src + t * 4, dst + t * 4);
        }
        // stage V block transposed: Vts[d][key]
#pragma unroll
        for (int rep = 0; rep < 8; ++rep) {
            const int linear = rep * 128 + tid;      // 1024 groups of 8 d-values
            const int kk = linear >> 4;              // key in block
            const int dg = (linear & 15) * 8;        // d group base
            uint4 vv = *(const uint4*)(Vg + (size_t)(k0 + kk) * HID + hoff + dg);
            const bf16_t* pv = (const bf16_t*)&vv;
#pragma unroll
            for (int e = 0; e < 8; ++e) Vts[(dg + e) * LDV + kk] = pv[e];
        }
        wait_async<0>();
        __syncthreads();

        // scores: S(16q x 64k) = Q(16x128) * K^T
        v8f sacc[4];
#pragma unroll
        for (int nf = 0; nf < 4; ++nf) sacc[nf] = vzero8();
#pragma unroll
        for (int nf = 0; nf < 4; ++nf)
#pragma unroll
            for (int kc = 0; kc < 4; ++kc)
                sacc[nf] = wmma_bf16f32(
                    qf[kc], frag_ld(Ks + nf * 16 * LDK + kc * 32, LDK, lane), sacc[nf]);

        // online softmax: row m = j + 8*hh lives in one 16-lane half-group
#pragma unroll
        for (int j = 0; j < 8; ++j) {
            const int mr = wave * 16 + hh * 8 + j;   // row within 64-row tile
            float s[4];
#pragma unroll
            for (int f = 0; f < 4; ++f)
                s[f] = sacc[f][j] * kscale + Ms[mr * LDM + f * 16 + ln];
            float mx = fmaxf(fmaxf(s[0], s[1]), fmaxf(s[2], s[3]));
#pragma unroll
            for (int off = 1; off < 16; off <<= 1)
                mx = fmaxf(mx, __shfl_xor(mx, off, 32));
            const float mnew = fmaxf(mrow[j], mx);
            const float corr = exp2f((mrow[j] - mnew) * log2e);
            float rs = 0.f;
#pragma unroll
            for (int f = 0; f < 4; ++f) {
                const float p = exp2f((s[f] - mnew) * log2e);
                sacc[f][j] = p;
                rs += p;
            }
#pragma unroll
            for (int off = 1; off < 16; off <<= 1)
                rs += __shfl_xor(rs, off, 32);
            lrow[j] = lrow[j] * corr + rs;
            mrow[j] = mnew;
#pragma unroll
            for (int nf = 0; nf < 8; ++nf) oacc[nf][j] *= corr;
        }

        // P -> wave-private LDS (reshape C/D layout into A-fragment layout)
#pragma unroll
        for (int f = 0; f < 4; ++f)
#pragma unroll
            for (int j = 0; j < 8; ++j)
                Pw[(hh * 8 + j) * LDV + f * 16 + ln] = (bf16_t)sacc[f][j];
        // wave-private buffer; DS ops from the same wave are in order.

        // O += P(16x64) * V(64x128)
        const v16bf pf0 = frag_ld(Pw, LDV, lane);
        const v16bf pf1 = frag_ld(Pw + 32, LDV, lane);
#pragma unroll
        for (int nf = 0; nf < 8; ++nf) {
            const v16bf vf0 = frag_ld(Vts + nf * 16 * LDV, LDV, lane);
            const v16bf vf1 = frag_ld(Vts + nf * 16 * LDV + 32, LDV, lane);
            oacc[nf] = wmma_bf16f32(pf0, vf0, oacc[nf]);
            oacc[nf] = wmma_bf16f32(pf1, vf1, oacc[nf]);
        }
    }

    float linv[8];
#pragma unroll
    for (int j = 0; j < 8; ++j) linv[j] = 1.0f / lrow[j];
#pragma unroll
    for (int nf = 0; nf < 8; ++nf)
#pragma unroll
        for (int j = 0; j < 8; ++j) {
            const int qm = q0 + hh * 8 + j;
            O[(size_t)qm * HID + hoff + nf * 16 + ln] =
                (bf16_t)(oacc[nf][j] * linv[j]);
        }
}

// ---------------- host-side orchestration ----------------
extern "C" void kernel_launch(void* const* d_in, const int* in_sizes, int n_in,
                              void* d_out, int out_size, void* d_ws, size_t ws_size,
                              hipStream_t stream) {
    (void)in_sizes; (void)n_in; (void)out_size; (void)ws_size;

    const float* hs   = (const float*)d_in[0];
    const float* mask = (const float*)d_in[1];
    const float* qV   = (const float*)d_in[2];
    const float* qU   = (const float*)d_in[3];
    const float* kV   = (const float*)d_in[4];
    const float* kU   = (const float*)d_in[5];
    const float* vV   = (const float*)d_in[6];
    const float* vU   = (const float*)d_in[7];
    const float* oW   = (const float*)d_in[8];
    const float* ob   = (const float*)d_in[9];

    // workspace carve-up (bf16 elements), total ~120 MB
    bf16_t* w = (bf16_t*)d_ws;
    bf16_t* Xb  = w; w += (size_t)S_LEN * HID;
    bf16_t* qVb = w; w += (size_t)RANK * HID;
    bf16_t* qUb = w; w += (size_t)HID * RANK;
    bf16_t* kVb = w; w += (size_t)RANK * HID;
    bf16_t* kUb = w; w += (size_t)HID * RANK;
    bf16_t* vVb = w; w += (size_t)RANK * HID;
    bf16_t* vUb = w; w += (size_t)HID * RANK;
    bf16_t* oWb = w; w += (size_t)HID * HID;
    bf16_t* Tb  = w; w += (size_t)S_LEN * RANK;
    bf16_t* Qb  = w; w += (size_t)S_LEN * HID;
    bf16_t* Kb  = w; w += (size_t)S_LEN * HID;
    bf16_t* Vb  = w; w += (size_t)S_LEN * HID;
    bf16_t* AOb = w; w += (size_t)S_LEN * HID;

    auto conv = [&](const float* s, bf16_t* d, size_t n) {
        conv_f32_bf16<<<dim3((unsigned)((n + 1023) / 1024)), dim3(256), 0, stream>>>(
            s, d, (int)n);
    };
    conv(hs, Xb, (size_t)S_LEN * HID);
    conv(qV, qVb, (size_t)RANK * HID);
    conv(qU, qUb, (size_t)HID * RANK);
    conv(kV, kVb, (size_t)RANK * HID);
    conv(kU, kUb, (size_t)HID * RANK);
    conv(vV, vVb, (size_t)RANK * HID);
    conv(vU, vUb, (size_t)HID * RANK);
    conv(oW, oWb, (size_t)HID * HID);

    const dim3 blk(128);
    const dim3 gVT(RANK / 128, S_LEN / 128);  // [4096,2048]x[1024,2048]^T
    const dim3 gUT(HID / 128, S_LEN / 128);   // [4096,1024]x[2048,1024]^T
    const dim3 gO(HID / 128, S_LEN / 128);    // [4096,2048]x[2048,2048]^T

    // Q
    gemm_bf16_wmma<true><<<gVT, blk, 0, stream>>>(Xb, qVb, Tb, nullptr, S_LEN, RANK, HID);
    gemm_bf16_wmma<true><<<gUT, blk, 0, stream>>>(Tb, qUb, Qb, nullptr, S_LEN, HID, RANK);
    // K
    gemm_bf16_wmma<true><<<gVT, blk, 0, stream>>>(Xb, kVb, Tb, nullptr, S_LEN, RANK, HID);
    gemm_bf16_wmma<true><<<gUT, blk, 0, stream>>>(Tb, kUb, Kb, nullptr, S_LEN, HID, RANK);
    // V
    gemm_bf16_wmma<true><<<gVT, blk, 0, stream>>>(Xb, vVb, Tb, nullptr, S_LEN, RANK, HID);
    gemm_bf16_wmma<true><<<gUT, blk, 0, stream>>>(Tb, vUb, Vb, nullptr, S_LEN, HID, RANK);

    // attention
    flash_attn_wmma<<<dim3(S_LEN / 64, HEADS), blk, 0, stream>>>(Qb, Kb, Vb, mask, AOb);

    // output projection + bias -> fp32 d_out
    gemm_bf16_wmma<false><<<gO, blk, 0, stream>>>(AOb, oWb, d_out, ob, S_LEN, HID, HID);
}
